// QPSpeedPlanning_42365557408286
// MI455X (gfx1250) — matrix operand order repaired and checked by reading, not compiled, not run on hardware
//
#include <hip/hip_runtime.h>
#include <hip/hip_bf16.h>

// ---------------------------------------------------------------------------
// Types for CDNA5 WMMA (gfx1250, wave32): bf16 16x16x32, f32 accumulate.
// ---------------------------------------------------------------------------
typedef __bf16 bf16_t;
typedef bf16_t v16bf __attribute__((ext_vector_type(16)));
typedef unsigned short v16us __attribute__((ext_vector_type(16)));
typedef float v8f __attribute__((ext_vector_type(8)));

__device__ __forceinline__ unsigned short f2bf(float f) {
    unsigned int u = __float_as_uint(f);
    unsigned int r = 0x7FFFu + ((u >> 16) & 1u);   // round-to-nearest-even
    return (unsigned short)((u + r) >> 16);
}

// A-matrix (16x32, 16-bit) element K for VGPR v, pair p, lane-half hi
// (CDNA5 ISA §7.12.2)
__device__ __forceinline__ int a_kidx(int v, int p, int hi) {
    return ((v < 4) ? 0 : 16) + hi * 8 + (v & 3) * 2 + p;
}
// B-matrix (32x16, 16-bit) helper (unused), its own order: lanes 0-15 take K=0..15, lanes 16-31 K=16..31
// => fragment element e corresponds to K = hi*16 + e (contiguous!)
__device__ __forceinline__ int b_kidx(int v, int p, int hi) {
    return hi * 16 + v * 2 + p;
}

// ---------------------------------------------------------------------------
// Kernel 1: conv1 (3->32ch, 3x3, valid) + bias + relu + 2x2 maxpool, fused.
// Implicit GEMM: M=32 (ch), K=27 pad 32. Lane column = POOLED output column;
// 4 B-frags per tile (y0/y1 x x-even/x-odd) -> pooling is in-lane max.
// Output: pooled1 bf16 NHWC (B, 26, 26, 32) -> 16B packed store per lane.
// ---------------------------------------------------------------------------
#define IMG_LDS 11712   // 4*54*54 padded (ch3 = zero pad for K=27..31 + OOB slack)

__global__ void __launch_bounds__(256)
conv1_kernel(const float* __restrict__ x, const float* __restrict__ w,
             const float* __restrict__ bias, unsigned short* __restrict__ out)
{
    __shared__ unsigned short img_l[IMG_LDS];
    __shared__ unsigned short w1[32 * 32];
    __shared__ int d1[32];
    __shared__ float b1_l[32];

    const int b   = blockIdx.x;
    const int tid = threadIdx.x;
    const size_t xbase = (size_t)b * 8751;

    for (int i = tid; i < IMG_LDS; i += 256)
        img_l[i] = (i < 8748) ? f2bf(x[xbase + i]) : (unsigned short)0;
    for (int i = tid; i < 1024; i += 256) {
        int ch = i >> 5, K = i & 31;
        w1[i] = (K < 27) ? f2bf(w[ch * 27 + K]) : (unsigned short)0;
    }
    if (tid < 32) {
        int c = tid / 9, r = tid % 9;
        d1[tid] = c * 2916 + (r / 3) * 54 + (r % 3);
        b1_l[tid] = bias[tid];
    }
    __syncthreads();

    const int lane = tid & 31, wv = tid >> 5;
    const int n = lane & 15, hi = lane >> 4;

    // cache both 16-channel A fragments (weights) once per wave
    v16bf afr[2];
#pragma unroll
    for (int mt = 0; mt < 2; ++mt) {
        v16us au;
#pragma unroll
        for (int e = 0; e < 16; ++e) {
            int v = e >> 1, p = e & 1;
            au[e] = w1[(mt * 16 + n) * 32 + a_kidx(v, p, hi)];
        }
        afr[mt] = __builtin_bit_cast(v16bf, au);
    }

    // 104 tiles: (row-pair 0..25) x (pooled-x-tile 0..1) x (ch-tile 0..1)
    for (int t = wv; t < 104; t += 8) {
        int rp = t >> 2, xt = (t >> 1) & 1, mt = t & 1;
        int y0 = rp * 2;
        int xe = xt * 32 + 2 * n;               // conv column of x-even frag

        v16us bu00, bu01, bu10, bu11;
#pragma unroll
        for (int e = 0; e < 16; ++e) {
            int v = e >> 1, p = e & 1;
            int a0 = y0 * 54 + xe + d1[a_kidx(v, p, hi)];
            bu00[e] = img_l[a0];
            bu01[e] = img_l[a0 + 1];
            bu10[e] = img_l[a0 + 54];
            bu11[e] = img_l[a0 + 55];
        }
        v16bf f00 = __builtin_bit_cast(v16bf, bu00);
        v16bf f01 = __builtin_bit_cast(v16bf, bu01);
        v16bf f10 = __builtin_bit_cast(v16bf, bu10);
        v16bf f11 = __builtin_bit_cast(v16bf, bu11);

        v8f c00 = {}, c01 = {}, c10 = {}, c11 = {};
        c00 = __builtin_amdgcn_wmma_f32_16x16x32_bf16(false, afr[mt], false, f00,
                                                      (short)0, c00, false, false);
        c01 = __builtin_amdgcn_wmma_f32_16x16x32_bf16(false, afr[mt], false, f01,
                                                      (short)0, c01, false, false);
        c10 = __builtin_amdgcn_wmma_f32_16x16x32_bf16(false, afr[mt], false, f10,
                                                      (short)0, c10, false, false);
        c11 = __builtin_amdgcn_wmma_f32_16x16x32_bf16(false, afr[mt], false, f11,
                                                      (short)0, c11, false, false);

        int px = xt * 16 + n;                    // pooled output column
        // pack 8 consecutive output channels (base + v) into one 16B store
        unsigned int pk[4];
#pragma unroll
        for (int v2 = 0; v2 < 4; ++v2) {
            int ch0 = mt * 16 + hi * 8 + 2 * v2;
            float m0 = fmaxf(fmaxf(c00[2*v2],   c01[2*v2]),   fmaxf(c10[2*v2],   c11[2*v2]));
            float m1 = fmaxf(fmaxf(c00[2*v2+1], c01[2*v2+1]), fmaxf(c10[2*v2+1], c11[2*v2+1]));
            m0 = fmaxf(m0 + b1_l[ch0],     0.f);
            m1 = fmaxf(m1 + b1_l[ch0 + 1], 0.f);
            pk[v2] = (unsigned int)f2bf(m0) | ((unsigned int)f2bf(m1) << 16);
        }
        if (px < 26) {
            size_t idx = ((size_t)b * 676 + rp * 26 + px) * 32 + mt * 16 + hi * 8;
            uint4 u4; u4.x = pk[0]; u4.y = pk[1]; u4.z = pk[2]; u4.w = pk[3];
            *reinterpret_cast<uint4*>(out + idx) = u4;
        }
    }
}

// ---------------------------------------------------------------------------
// Kernel 2: conv2 (32->64ch, 3x3, valid) + bias + relu + 2x2 maxpool, fused.
// NHWC input -> 9 per-tap GEMMs, each K = 32 channels: B fragments are two
// contiguous ds_load_b128 per fragment (no gather, no packing movs).
// Input staged into LDS via GLOBAL_LOAD_ASYNC_TO_LDS_B128 (ASYNCcnt DMA).
// Output: pooled2 f32 NHWC (B, 12, 12, 64).
// ---------------------------------------------------------------------------
#define IN2_ELEMS 21632   // 26*26*32 bf16, NHWC, no slack needed

__global__ void __launch_bounds__(256)
conv2_kernel(const unsigned short* __restrict__ pooled1,
             const float* __restrict__ w, const float* __restrict__ bias,
             float* __restrict__ out)
{
    __shared__ __align__(64) unsigned short in_l[IN2_ELEMS];
    __shared__ float b2_l[64];

    const int b   = blockIdx.x;
    const int tid = threadIdx.x;
    const size_t ibase = (size_t)b * IN2_ELEMS;

    // --- async DMA stage: 21632 bf16 = 43264 B = 2704 x 16B chunks ----------
    {
        const unsigned short* gsrc = pooled1 + ibase;
        unsigned lds_base = (unsigned)(uintptr_t)(&in_l[0]);  // LDS offset in low 32b
        for (int i = tid; i < 2704; i += 256) {
            unsigned goff = (unsigned)i * 16u;
            unsigned ldsa = lds_base + goff;
            asm volatile("global_load_async_to_lds_b128 %0, %1, %2 offset:0"
                         :: "v"(ldsa), "v"(goff), "s"(gsrc)
                         : "memory");
        }
        if (tid < 64) b2_l[tid] = bias[tid];
        asm volatile("s_wait_asynccnt 0" ::: "memory");
    }
    __syncthreads();

    const int lane = tid & 31, wv = tid >> 5;
    const int n = lane & 15, hi = lane >> 4;
    const int mt = wv & 3, half = wv >> 2;

    // per-tap A fragments: row = out-channel, K = in-channel c
    // w2 layout OIHW: w[o*288 + c*9 + tap]
    v16bf afr[9];
#pragma unroll
    for (int t9 = 0; t9 < 9; ++t9) {
        v16us au;
#pragma unroll
        for (int e = 0; e < 16; ++e) {
            int v = e >> 1, p = e & 1;
            au[e] = f2bf(w[(mt * 16 + n) * 288 + (hi * 16 + e) * 9 + t9]);
        }
        afr[t9] = __builtin_bit_cast(v16bf, au);
    }

    // 12 row-pair tiles; lane column n = pooled output column (12 valid of 16)
    for (int rp = half; rp < 12; rp += 2) {
        int y0 = rp * 2;
        int xe = 2 * n;                          // conv column of x-even frag
        v8f c00 = {}, c01 = {}, c10 = {}, c11 = {};
#pragma unroll
        for (int dy = 0; dy < 3; ++dy) {
#pragma unroll
            for (int dx = 0; dx < 3; ++dx) {
                int t9 = dy * 3 + dx;
                int p00 = (y0 + dy) * 26 + (xe + dx);   // pixel index, NHWC
                // fragment = 16 contiguous channels at one pixel
                v16us bu00 = *reinterpret_cast<const v16us*>(&in_l[p00 * 32 + hi * 16]);
                v16us bu01 = *reinterpret_cast<const v16us*>(&in_l[(p00 + 1) * 32 + hi * 16]);
                v16us bu10 = *reinterpret_cast<const v16us*>(&in_l[(p00 + 26) * 32 + hi * 16]);
                v16us bu11 = *reinterpret_cast<const v16us*>(&in_l[(p00 + 27) * 32 + hi * 16]);
                v16bf f00 = __builtin_bit_cast(v16bf, bu00);
                v16bf f01 = __builtin_bit_cast(v16bf, bu01);
                v16bf f10 = __builtin_bit_cast(v16bf, bu10);
                v16bf f11 = __builtin_bit_cast(v16bf, bu11);
                c00 = __builtin_amdgcn_wmma_f32_16x16x32_bf16(false, afr[t9], false, f00,
                                                              (short)0, c00, false, false);
                c01 = __builtin_amdgcn_wmma_f32_16x16x32_bf16(false, afr[t9], false, f01,
                                                              (short)0, c01, false, false);
                c10 = __builtin_amdgcn_wmma_f32_16x16x32_bf16(false, afr[t9], false, f10,
                                                              (short)0, c10, false, false);
                c11 = __builtin_amdgcn_wmma_f32_16x16x32_bf16(false, afr[t9], false, f11,
                                                              (short)0, c11, false, false);
            }
        }
        // epilogue: bias + relu + pooled max; 8 consecutive channels per lane
        float m[8];
#pragma unroll
        for (int v = 0; v < 8; ++v) {
            int ch = mt * 16 + hi * 8 + v;
            float mm = fmaxf(fmaxf(c00[v], c01[v]), fmaxf(c10[v], c11[v]));
            m[v] = fmaxf(mm + b2_l[ch], 0.f);
        }
        if (n < 12) {
            size_t idx = ((size_t)b * 144 + rp * 12 + n) * 64 + mt * 16 + hi * 8;
            float4 s0 = make_float4(m[0], m[1], m[2], m[3]);
            float4 s1 = make_float4(m[4], m[5], m[6], m[7]);
            *reinterpret_cast<float4*>(out + idx)     = s0;
            *reinterpret_cast<float4*>(out + idx + 4) = s1;
        }
    }
}

// ---------------------------------------------------------------------------
// One-shot: permute fc weights to NHWC activation order so the head kernel's
// reads stay fully coalesced.  fwT[j][pix*64+ch] = fw[j][ch*144+pix]
// ---------------------------------------------------------------------------
__global__ void __launch_bounds__(256)
permw_kernel(const float* __restrict__ fw, float* __restrict__ fwT)
{
    int i = blockIdx.x * 256 + threadIdx.x;
    if (i >= 6 * 9219) return;
    int j = i / 9219, ii = i % 9219;
    int src;
    if (ii < 9216) {
        int ch = ii & 63, pix = ii >> 6;
        src = j * 9219 + ch * 144 + pix;
    } else {
        src = i;                                  // speed / dest / car tail
    }
    fwT[i] = fw[src];
}

// ---------------------------------------------------------------------------
// Kernel 3: fc (9219 -> 6) + relu, then 2x2 SVD clip + 4x4 KKT solve.
// ---------------------------------------------------------------------------
__global__ void __launch_bounds__(256)
head_kernel(const float* __restrict__ pooled2, const float* __restrict__ x,
            const float* __restrict__ fwT, const float* __restrict__ fb,
            float* __restrict__ out)
{
    __shared__ float red[8][6];
    const int b = blockIdx.x;
    const int tid = threadIdx.x;
    float acc[6] = {0.f, 0.f, 0.f, 0.f, 0.f, 0.f};

    const float* h = pooled2 + (size_t)b * 9216;
    for (int i = tid; i < 9216; i += 256) {
        float hv = h[i];
#pragma unroll
        for (int j = 0; j < 6; ++j) acc[j] += hv * fwT[j * 9219 + i];
    }
    if (tid < 3) {   // speed, dest_distance, car_distance
        float hv = x[(size_t)b * 8751 + 8748 + tid];
#pragma unroll
        for (int j = 0; j < 6; ++j) acc[j] += hv * fwT[j * 9219 + 9216 + tid];
    }
    const int lane = tid & 31, wv = tid >> 5;
#pragma unroll
    for (int j = 0; j < 6; ++j)
        for (int off = 16; off > 0; off >>= 1)
            acc[j] += __shfl_down(acc[j], off, 32);
    if (lane == 0)
        for (int j = 0; j < 6; ++j) red[wv][j] = acc[j];
    __syncthreads();

    if (tid == 0) {
        float y[6];
        for (int j = 0; j < 6; ++j) {
            float s = fb[j];
            for (int k = 0; k < 8; ++k) s += red[k][j];
            y[j] = fmaxf(s, 0.f);
        }
        // closed-form 2x2 SVD of A = [[y0,y1],[y2,y3]]: A = R(gam) diag(s1,s2s) R(beta)
        float aa = y[0], ab = y[1], ac = y[2], ad = y[3];
        float E = 0.5f * (aa + ad), F = 0.5f * (aa - ad);
        float G = 0.5f * (ac + ab), H = 0.5f * (ac - ab);
        float Qh = sqrtf(E * E + H * H), Rh = sqrtf(F * F + G * G);
        float s1 = Qh + Rh, s2s = Qh - Rh;
        float sg2 = (s2s < 0.f) ? -1.f : 1.f;
        float s2 = fabsf(s2s);
        float a1 = atan2f(G, F), a2 = atan2f(H, E);
        float beta = 0.5f * (a2 - a1), gam = 0.5f * (a2 + a1);
        float cg = cosf(gam), sg = sinf(gam);
        float cb = cosf(beta), sb = sinf(beta);
        // clip: smax = s1 >= s2 >= 0; EPSILON = 10
        float s1p = s1, s2p = fmaxf(s2, s1 * 0.1f);
        if (s1p <= 0.f) { s1p = 0.01f; s2p = 0.01f; }
        // Vh = [[cb,-sb],[sg2*sb, sg2*cb]];  A_p = U * diag(s1p,s2p) * Vh
        float m00 = s1p * cb,        m01 = -s1p * sb;
        float m10 = s2p * sg2 * sb,  m11 = s2p * sg2 * cb;
        float Ap00 = cg * m00 - sg * m10, Ap01 = cg * m01 - sg * m11;
        float Ap10 = sg * m00 + cg * m10, Ap11 = sg * m01 + cg * m11;
        // KKT = [[Q, Ap^T],[Ap, 0]], rhs = [-p, b] = [0, 1000, y4, y5]
        float Kt[4][5] = {
            {1.f,  0.f,  Ap00, Ap10, 0.f},
            {0.f,  0.5f, Ap01, Ap11, 1000.f},
            {Ap00, Ap01, 0.f,  0.f,  y[4]},
            {Ap10, Ap11, 0.f,  0.f,  y[5]}
        };
        for (int col = 0; col < 4; ++col) {
            int piv = col; float best = fabsf(Kt[col][col]);
            for (int r = col + 1; r < 4; ++r) {
                float vv = fabsf(Kt[r][col]);
                if (vv > best) { best = vv; piv = r; }
            }
            if (piv != col)
                for (int cc = col; cc < 5; ++cc) {
                    float t = Kt[col][cc]; Kt[col][cc] = Kt[piv][cc]; Kt[piv][cc] = t;
                }
            float inv = 1.f / Kt[col][col];
            for (int r = col + 1; r < 4; ++r) {
                float f = Kt[r][col] * inv;
                for (int cc = col; cc < 5; ++cc) Kt[r][cc] -= f * Kt[col][cc];
            }
        }
        float sol[4];
        for (int r = 3; r >= 0; --r) {
            float s = Kt[r][4];
            for (int cc = r + 1; cc < 4; ++cc) s -= Kt[r][cc] * sol[cc];
            sol[r] = s / Kt[r][r];
        }
        out[b * 2 + 0] = sol[0];
        out[b * 2 + 1] = sol[1];
    }
}

// ---------------------------------------------------------------------------
extern "C" void kernel_launch(void* const* d_in, const int* in_sizes, int n_in,
                              void* d_out, int out_size, void* d_ws, size_t ws_size,
                              hipStream_t stream)
{
    (void)n_in; (void)out_size; (void)ws_size;
    const float* x  = (const float*)d_in[0];
    const float* w1 = (const float*)d_in[1];
    const float* b1 = (const float*)d_in[2];
    const float* w2 = (const float*)d_in[3];
    const float* b2 = (const float*)d_in[4];
    const float* fw = (const float*)d_in[5];
    const float* fb = (const float*)d_in[6];
    float* out = (float*)d_out;
    const int B = in_sizes[0] / 8751;

    unsigned short* pooled1 = (unsigned short*)d_ws;                   // B*26*26*32 bf16 NHWC
    float* pooled2 = (float*)((char*)d_ws + (size_t)B * 21632 * 2);    // B*12*12*64 f32 NHWC
    float* fwT     = (float*)((char*)d_ws + (size_t)B * 21632 * 2
                                          + (size_t)B * 9216 * 4);     // 6*9219 f32

    permw_kernel<<<(6 * 9219 + 255) / 256, 256, 0, stream>>>(fw, fwT);
    conv1_kernel<<<B, 256, 0, stream>>>(x, w1, b1, pooled1);
    conv2_kernel<<<B, 256, 0, stream>>>(pooled1, w2, b2, pooled2);
    head_kernel<<<B, 256, 0, stream>>>(pooled2, x, fwT, fb, out);
}
